// CustomAttentionLayer_49993419325883
// MI455X (gfx1250) — matrix operand order repaired and checked by reading, not compiled, not run on hardware
//
#include <hip/hip_runtime.h>
#include <hip/hip_bf16.h>

// CDNA5 (gfx1250) wave32 WMMA / TDM types
typedef __attribute__((ext_vector_type(16))) __bf16      v16bf;
typedef __attribute__((ext_vector_type(8)))  __bf16      v8bf;
typedef __attribute__((ext_vector_type(8)))  float       v8f;
typedef __attribute__((ext_vector_type(4)))  unsigned int v4u;
typedef __attribute__((ext_vector_type(8)))  int         v8i;
typedef __attribute__((ext_vector_type(4)))  int         v4i;

// ---------------------------------------------------------------------------
// Tensor Data Mover: async-load a 2D tile [trows x tcols] of bf16 (row-major,
// row stride = ld elements) from global memory into LDS at lds_off.
// Issued once per calling wave; completion tracked with TENSORcnt.
// D# layout per CDNA5 ISA (group0 128b, group1 256b).
// ---------------------------------------------------------------------------
__device__ __forceinline__
void tdm_load_tile_bf16(const __bf16* gptr, unsigned lds_off,
                        int trows, int tcols, int ld, int tensor_rows) {
    unsigned long long ga = (unsigned long long)(uintptr_t)(const void*)gptr;
    v4u g0;
    g0[0] = 1u;                                   // count=1, no gather, user mode
    g0[1] = lds_off;                              // LDS byte address
    g0[2] = (unsigned)(ga & 0xffffffffull);       // global_addr[31:0]
    g0[3] = (unsigned)((ga >> 32) & 0x1ffffffu)   // global_addr[56:32]
          | (2u << 30);                           // type=2 ("image")
    v8i g1;
    g1[0] = (1 << 16);                            // wg_mask=0, data_size=1 (2B)
    g1[1] = (ld & 0xffff) << 16;                  // tensor_dim0[15:0]
    g1[2] = ((ld >> 16) & 0xffff)                 // tensor_dim0[31:16]
          | ((tensor_rows & 0xffff) << 16);       // tensor_dim1[15:0]
    g1[3] = ((tensor_rows >> 16) & 0xffff)        // tensor_dim1[31:16]
          | ((tcols & 0xffff) << 16);             // tile_dim0
    g1[4] = trows & 0xffff;                       // tile_dim1 (tile_dim2 = 0)
    g1[5] = ld;                                   // tensor_dim0_stride[31:0]
    g1[6] = 0;                                    // stride[47:32] | dim1_stride lo
    g1[7] = 0;
    v4i g2 = {0, 0, 0, 0};
    v4i g3 = {0, 0, 0, 0};
#if __clang_major__ >= 23
    v8i g4 = {0, 0, 0, 0, 0, 0, 0, 0};
    __builtin_amdgcn_tensor_load_to_lds(g0, g1, g2, g3, g4, 0);
#else
    __builtin_amdgcn_tensor_load_to_lds(g0, g1, g2, g3, 0);
#endif
}

// ---------------------------------------------------------------------------
// Elementwise cast f32 -> bf16 (4 elems/thread)
// ---------------------------------------------------------------------------
__global__ __launch_bounds__(256)
void cast_f32_bf16(const float* __restrict__ in, __bf16* __restrict__ out, long n) {
    long idx = ((long)blockIdx.x * blockDim.x + threadIdx.x) * 4;
    if (idx + 3 < n) {
        float4 v = *(const float4*)(in + idx);
        out[idx + 0] = (__bf16)v.x;
        out[idx + 1] = (__bf16)v.y;
        out[idx + 2] = (__bf16)v.z;
        out[idx + 3] = (__bf16)v.w;
    }
}

// ---------------------------------------------------------------------------
// Rotary over full hidden dim (d = 1024, H = 2048), interleaved, f32 -> bf16
// grid: (1024/256, rows)
// ---------------------------------------------------------------------------
__global__ __launch_bounds__(256)
void rotary_kernel(const float* __restrict__ x, const float* __restrict__ ang,
                   __bf16* __restrict__ out) {
    int  i   = blockIdx.x * blockDim.x + threadIdx.x;   // 0..1023
    long row = blockIdx.y;
    float a = ang[i];
    float c, s;
    __sincosf(a, &s, &c);
    const float* xr = x + row * 2048;
    float vr = xr[i];
    float vi = xr[i + 1024];
    __bf16* o = out + row * 2048;
    o[2 * i + 0] = (__bf16)(vr * c - vi * s);
    o[2 * i + 1] = (__bf16)(vr * s + vi * c);
}

// ---------------------------------------------------------------------------
// Tiled transpose + cast: f32 [b, S, H] -> bf16 [b, H, S]
// block (32,8), grid (H/32, S/32, B)
// ---------------------------------------------------------------------------
__global__ __launch_bounds__(256)
void transpose_cast(const float* __restrict__ in, __bf16* __restrict__ out,
                    int S, int Hd) {
    __shared__ float tile[32][33];
    long base = (long)blockIdx.z * S * Hd;
    int  s0 = blockIdx.y * 32, h0 = blockIdx.x * 32;
    int  tx = threadIdx.x, ty = threadIdx.y;
    #pragma unroll
    for (int r = 0; r < 32; r += 8)
        tile[ty + r][tx] = in[base + (long)(s0 + ty + r) * Hd + h0 + tx];
    __syncthreads();
    #pragma unroll
    for (int r = 0; r < 32; r += 8)
        out[base + (long)(h0 + ty + r) * S + s0 + tx] = (__bf16)tile[tx][ty + r];
}

// ---------------------------------------------------------------------------
// Row softmax: f32 scores in, bf16 probs out. One 256-thread block per row.
// ---------------------------------------------------------------------------
__global__ __launch_bounds__(256)
void softmax_kernel(const float* __restrict__ sc, __bf16* __restrict__ pr, int ncol) {
    long row = blockIdx.x;
    const float* r = sc + row * (long)ncol;
    __bf16*      o = pr + row * (long)ncol;
    __shared__ float red[256];
    int tid = threadIdx.x;

    float mx = -3.0e38f;
    for (int c = tid; c < ncol; c += 256) mx = fmaxf(mx, r[c]);
    red[tid] = mx; __syncthreads();
    for (int s = 128; s > 0; s >>= 1) {
        if (tid < s) red[tid] = fmaxf(red[tid], red[tid + s]);
        __syncthreads();
    }
    mx = red[0]; __syncthreads();

    float sum = 0.0f;
    for (int c = tid; c < ncol; c += 256) sum += __expf(r[c] - mx);
    red[tid] = sum; __syncthreads();
    for (int s = 128; s > 0; s >>= 1) {
        if (tid < s) red[tid] += red[tid + s];
        __syncthreads();
    }
    float inv = 1.0f / red[0];
    for (int c = tid; c < ncol; c += 256) o[c] = (__bf16)(__expf(r[c] - mx) * inv);
}

// ---------------------------------------------------------------------------
// bf16 WMMA GEMM (NT form): C[M,N] = alpha * A[M,K] @ B[N,K]^T, f32 output.
//  - Block tile 128(M) x 256(N), BK = 64 (two WMMA K-substeps per stage)
//  - 8 waves in 2x4; wave tile 64x64 = 4x4 fragments (16 wmma / substep)
//  - Global->LDS staging via Tensor Data Mover, double buffered:
//    wave 0 issues TDM loads for stage s+1, waits TENSORcnt<=2 (stage s is
//    the oldest pair, TDM completes in-order per wave), workgroup barrier
//    releases compute waves.
//  - Batched via blockIdx.z with element strides sA/sB/sC.
// Requires: M%128==0, N%256==0, K%64==0 (true for all calls here).
// ---------------------------------------------------------------------------
#define TBM 128
#define TBN 256
#define TBK 64

__global__ __launch_bounds__(256)
void gemm_bf16_nt(const __bf16* __restrict__ A, const __bf16* __restrict__ B,
                  float* __restrict__ C, int M, int N, int K,
                  long sA, long sB, long sC, float alpha) {
    __shared__ __bf16 As[2][TBM * TBK];   // 2 x 16 KB
    __shared__ __bf16 Bs[2][TBN * TBK];   // 2 x 32 KB

    const int bz = blockIdx.z;
    A += (long)bz * sA;
    B += (long)bz * sB;
    C += (long)bz * sC;

    const int m0   = blockIdx.y * TBM;
    const int n0   = blockIdx.x * TBN;
    const int tid  = threadIdx.x;
    const int lane = tid & 31;
    const int wave = tid >> 5;
    const int wm0  = (wave >> 2) * 64;        // wave M offset: 0 or 64
    const int wn0  = (wave & 3) * 64;         // wave N offset: 0/64/128/192
    const int l15  = lane & 15;
    const int kofA = (lane & 16) ? 16 : 0;    // A frag: 16 contiguous K values
    const int kofB = (lane & 16) ? 16 : 0;    // B frag: 16 contiguous K values

    const __bf16* Abase = A + (long)m0 * K;   // tile origin rows
    const __bf16* Bbase = B + (long)n0 * K;
    const unsigned ldsA0 = (unsigned)(uintptr_t)(void*)&As[0][0];
    const unsigned ldsA1 = (unsigned)(uintptr_t)(void*)&As[1][0];
    const unsigned ldsB0 = (unsigned)(uintptr_t)(void*)&Bs[0][0];
    const unsigned ldsB1 = (unsigned)(uintptr_t)(void*)&Bs[1][0];

    const int NK = K / TBK;

    // Preload stage 0 (wave 0 only; TDM ignores EXEC, branch is wave-uniform)
    if (wave == 0) {
        tdm_load_tile_bf16(Abase, ldsA0, TBM, TBK, K, M);
        tdm_load_tile_bf16(Bbase, ldsB0, TBN, TBK, K, N);
    }

    v8f acc[4][4] = {};

    for (int s = 0; s < NK; ++s) {
        if (wave == 0) {
            if (s + 1 < NK) {
                const int kt1 = (s + 1) * TBK;
                const unsigned la = ((s + 1) & 1) ? ldsA1 : ldsA0;
                const unsigned lb = ((s + 1) & 1) ? ldsB1 : ldsB0;
                tdm_load_tile_bf16(Abase + kt1, la, TBM, TBK, K, M);
                tdm_load_tile_bf16(Bbase + kt1, lb, TBN, TBK, K, N);
                __builtin_amdgcn_s_wait_tensorcnt(2);   // stage s (oldest 2) done
            } else {
                __builtin_amdgcn_s_wait_tensorcnt(0);   // last stage done
            }
        }
        __syncthreads();   // release compute waves; also fences prior LDS reads

        const __bf16* as = &As[s & 1][0];
        const __bf16* bs = &Bs[s & 1][0];

        #pragma unroll
        for (int ks = 0; ks < TBK; ks += 32) {
            v16bf bf[4];
            #pragma unroll
            for (int j = 0; j < 4; ++j)
                bf[j] = *(const v16bf*)&bs[(wn0 + 16 * j + l15) * TBK + ks + kofB];
            #pragma unroll
            for (int i = 0; i < 4; ++i) {
                const __bf16* ar = &as[(wm0 + 16 * i + l15) * TBK + ks];
                v8bf alo = *(const v8bf*)(ar + kofA);
                v8bf ahi = *(const v8bf*)(ar + 8 + kofA);
                v16bf af = __builtin_shufflevector(alo, ahi,
                             0, 1, 2, 3, 4, 5, 6, 7, 8, 9, 10, 11, 12, 13, 14, 15);
                #pragma unroll
                for (int j = 0; j < 4; ++j)
                    acc[i][j] = __builtin_amdgcn_wmma_f32_16x16x32_bf16(
                        false, af, false, bf[j], (short)0, acc[i][j], false, false);
            }
        }
        __syncthreads();   // stage fully consumed before TDM overwrites it
    }

    // Epilogue: D frag element e -> row (base + e + (lane>=16 ? 8 : 0)), col l15
    const int mg = (lane & 16) ? 8 : 0;
    #pragma unroll
    for (int i = 0; i < 4; ++i) {
        #pragma unroll
        for (int j = 0; j < 4; ++j) {
            long n = n0 + wn0 + 16 * j + l15;
            #pragma unroll
            for (int e = 0; e < 8; ++e) {
                long m = m0 + wm0 + 16 * i + e + mg;
                C[m * (long)N + n] = acc[i][j][e] * alpha;
            }
        }
    }
}

static inline void run_gemm(const __bf16* A, const __bf16* B, float* C,
                            int M, int N, int K, int batch,
                            long sA, long sB, long sC, float alpha,
                            hipStream_t s) {
    dim3 grid(N / TBN, M / TBM, batch);
    gemm_bf16_nt<<<grid, dim3(256), 0, s>>>(A, B, C, M, N, K, sA, sB, sC, alpha);
}

// ---------------------------------------------------------------------------
// Orchestration
// ---------------------------------------------------------------------------
extern "C" void kernel_launch(void* const* d_in, const int* in_sizes, int n_in,
                              void* d_out, int out_size, void* d_ws, size_t ws_size,
                              hipStream_t stream) {
    const float* hs  = (const float*)d_in[0];   // [2,4096,2048] f32
    const float* ang = (const float*)d_in[1];   // [1024] f32
    const float* wq  = (const float*)d_in[2];   // [2048,2048] f32  (out, in)
    const float* wk  = (const float*)d_in[3];
    const float* wv  = (const float*)d_in[4];
    const float* wo  = (const float*)d_in[5];
    float* out = (float*)d_out;                 // [2,4096,2048] f32

    const int  B = 2, S = 4096, H = 2048;
    const long BSH = (long)B * S * H;           // 16,777,216
    const long HH  = (long)H * H;               //  4,194,304
    const long SS  = (long)S * S;               // 16,777,216
    const float scale = 0.08838834764831845f;   // 1/sqrt(H/num_heads) = 1/sqrt(128)

    char* ws = (char*)d_ws;
    __bf16* hsb    = (__bf16*)(ws + 0);             // 32 MB  bf16 hidden
    __bf16* wqb    = (__bf16*)(ws + 33554432L);     //  8 MB
    __bf16* wkb    = (__bf16*)(ws + 41943040L);     //  8 MB
    __bf16* wvb    = (__bf16*)(ws + 50331648L);     //  8 MB
    __bf16* wob    = (__bf16*)(ws + 58720256L);     //  8 MB
    __bf16* qrotb  = (__bf16*)(ws + 67108864L);     // 32 MB  bf16 rotated Q
    __bf16* krotb  = (__bf16*)(ws + 100663296L);    // 32 MB  bf16 rotated K
    __bf16* vtb    = (__bf16*)(ws + 134217728L);    // 32 MB  bf16 V^T [b,H,S]
    float*  f32s   = (float*) (ws + 167772160L);    // 64 MB  f32 scratch (reused)
    float*  scores = (float*) (ws + 234881024L);    // 128 MB f32 scores [b,S,S]
    __bf16* probsb = (__bf16*)(ws + 369098752L);    // 64 MB  bf16 probs [b,S,S]
    __bf16* attnb  = (__bf16*)(ws + 436207616L);    // 32 MB  bf16 attn out

    // ---- cast inputs to bf16 ----
    cast_f32_bf16<<<(int)(BSH / 4 / 256), 256, 0, stream>>>(hs, hsb, BSH);
    cast_f32_bf16<<<(int)(HH  / 4 / 256), 256, 0, stream>>>(wq, wqb, HH);
    cast_f32_bf16<<<(int)(HH  / 4 / 256), 256, 0, stream>>>(wk, wkb, HH);
    cast_f32_bf16<<<(int)(HH  / 4 / 256), 256, 0, stream>>>(wv, wvb, HH);
    cast_f32_bf16<<<(int)(HH  / 4 / 256), 256, 0, stream>>>(wo, wob, HH);

    // ---- Q = hs @ wq^T ; rotary -> bf16 ----
    run_gemm(hsb, wqb, f32s, B * S, H, H, 1, 0, 0, 0, 1.0f, stream);
    rotary_kernel<<<dim3(4, B * S), 256, 0, stream>>>(f32s, ang, qrotb);

    // ---- K = hs @ wk^T ; rotary -> bf16 ----
    run_gemm(hsb, wkb, f32s, B * S, H, H, 1, 0, 0, 0, 1.0f, stream);
    rotary_kernel<<<dim3(4, B * S), 256, 0, stream>>>(f32s, ang, krotb);

    // ---- V = hs @ wv^T ; transpose-cast -> V^T bf16 ----
    run_gemm(hsb, wvb, f32s, B * S, H, H, 1, 0, 0, 0, 1.0f, stream);
    transpose_cast<<<dim3(H / 32, S / 32, B), dim3(32, 8), 0, stream>>>(f32s, vtb, S, H);

    // ---- scores = scale * Qrot @ Krot^T  (batched over B) ----
    run_gemm(qrotb, krotb, scores, S, S, H, B,
             (long)S * H, (long)S * H, SS, scale, stream);

    // ---- probs = softmax(scores) -> bf16 ----
    softmax_kernel<<<B * S, 256, 0, stream>>>(scores, probsb, S);

    // ---- attn = probs @ V  ==  probs[M=S,K=S] @ (V^T)[N=H,K=S]^T ----
    run_gemm(probsb, vtb, f32s, S, H, S, B,
             SS, (long)H * S, (long)S * H, 1.0f, stream);

    // ---- cast attn to bf16, final projection: out = attn @ wo^T ----
    cast_f32_bf16<<<(int)(BSH / 4 / 256), 256, 0, stream>>>(f32s, attnb, BSH);
    run_gemm(attnb, wob, out, B * S, H, H, 1, 0, 0, 0, 1.0f, stream);
}
